// SoftPool_2d_30219389895197
// MI455X (gfx1250) — compile-verified, matrix operand order checked by reading, not run on hardware
//
#include <hip/hip_runtime.h>

typedef __attribute__((ext_vector_type(16))) _Float16 v16h;
typedef __attribute__((ext_vector_type(8)))  _Float16 v8h;
typedef __attribute__((ext_vector_type(2)))  _Float16 h2;
typedef __attribute__((ext_vector_type(8)))  float    v8f;

#define DIMD 64
#define HID  128
#define RR   128
#define CC   128
#define SRR  64
#define SCC  64
// windows = 32*64*64 = 131072 ; rows = 524288 ; 16-row tiles = 32768 ; 8 waves/block -> 4096 blocks

// Pre-swizzled B storage: for each (n, hi) group, 16 dwords in WMMA fragment order,
// padded to a 20-dword stride (16B aligned, breaks bank conflicts).
#define GRP_STRIDE_DW 20
#define LDSB_DW (256 * GRP_STRIDE_DW)   // 128 n * 2 hi groups -> 20 KB

// Branch-free transcendentals on the TRANS pipe (no libm saveexec regions).
static __device__ __forceinline__ float tanh_fast(float x) {
#if __has_builtin(__builtin_amdgcn_tanhf)
    return __builtin_amdgcn_tanhf(x);            // v_tanh_f32 (CDNA5 TRANS op)
#else
    const float e = __expf(2.0f * x);
    return 1.0f - 2.0f * __builtin_amdgcn_rcpf(e + 1.0f);
#endif
}
static __device__ __forceinline__ float sigmoid_fast(float x) {
    return __builtin_amdgcn_rcpf(1.0f + __expf(-x));
}

// 8 consecutive fp32 -> 8 f16 lanes of a v16h, via two b128 loads.
#define CVT8(dst, i0, ptr) do {                                     \
    const float4 _u = *(const float4*)(ptr);                        \
    const float4 _v = *(const float4*)((ptr) + 4);                  \
    dst[(i0) + 0] = (_Float16)_u.x; dst[(i0) + 1] = (_Float16)_u.y; \
    dst[(i0) + 2] = (_Float16)_u.z; dst[(i0) + 3] = (_Float16)_u.w; \
    dst[(i0) + 4] = (_Float16)_v.x; dst[(i0) + 5] = (_Float16)_v.y; \
    dst[(i0) + 6] = (_Float16)_v.z; dst[(i0) + 7] = (_Float16)_v.w; \
} while (0)

__global__ __launch_bounds__(256) void softpool2d_wmma_kernel(
    const float* __restrict__ x,    // [32][128][128][64]
    const float* __restrict__ Wx,   // [64][128]
    const float* __restrict__ bx,   // [128]
    const float* __restrict__ Wr,   // [128]
    const float* __restrict__ br,   // [1]
    float* __restrict__ out)        // [32][64][64][64]
{
    // W_expand in LDS, pre-swizzled into per-lane WMMA B-fragment order:
    // group g = n*2 + hi holds dwords u = c*8 + v, dword u = {W[kb][n], W[kb+1][n]},
    // kb = 32c + (v&4 ? 16 : 0) + 8*hi + 2*(v&3)  -- exactly the fragment layout.
    __shared__ __align__(16) h2 ldsB[LDSB_DW];

    const int tid = threadIdx.x;
    for (int i = tid; i < 256 * 16; i += 256) {
        const int g  = i >> 4;         // (n*2 + hi)
        const int u  = i & 15;
        const int n  = g >> 1;
        const int hi = g & 1;
        const int c  = u >> 3;
        const int v  = u & 7;
        const int kb = 32 * c + ((v & 4) ? 16 : 0) + 8 * hi + 2 * (v & 3);
        h2 pk;
        pk.x = (_Float16)Wx[(kb + 0) * HID + n];
        pk.y = (_Float16)Wx[(kb + 1) * HID + n];
        ldsB[g * GRP_STRIDE_DW + u] = pk;
    }
    __syncthreads();

    const v8h* ldsB8 = (const v8h*)ldsB;   // 4-dword quads; group stride = 5 quads

    const int  lane    = tid & 31;
    const int  wave    = tid >> 5;
    const int  tile    = blockIdx.x * 8 + wave;   // 16-row M tile
    const long winBase = (long)tile * 4;          // first of 4 windows owned by this wave

    // ---------- A fragments: lane holds row m = lane&15, K-half per lane>>4 ----------
    const int  m  = lane & 15;
    const int  hi = lane >> 4;                    // 0 -> K+0 , 1 -> K+8 within each 16-K half
    const long w  = winBase + (m >> 2);
    const int  we = m & 3;                        // window element (dr,dc) row-major
    const int  pw = (int)(w & 63);
    const int  qw = (int)((w >> 6) & 63);
    const int  bw = (int)(w >> 12);
    const float* rowA =
        x + (((long)bw * RR + (2 * qw + (we >> 1))) * CC + (2 * pw + (we & 1))) * DIMD;

    v16h a0, a1;   // K = 0..31 and K = 32..63 (per-lane runs are 8 floats contiguous)
    CVT8(a0, 0, rowA +      8 * hi);
    CVT8(a0, 8, rowA + 16 + 8 * hi);
    CVT8(a1, 0, rowA + 32 + 8 * hi);
    CVT8(a1, 8, rowA + 48 + 8 * hi);

    // ---------- gate GEMM over 8 N-tiles, folding in tanh + W_restore dot ----------
    float acc[8];
#pragma unroll
    for (int v = 0; v < 8; ++v) acc[v] = 0.0f;

    const int nl = lane & 15;                     // lane's column inside an N tile
#pragma unroll
    for (int t = 0; t < 8; ++t) {
        const int n = t * 16 + nl;
        const int g5 = (n * 2 + hi) * 5;          // quad index of this lane's group
        const v8h q0 = ldsB8[g5 + 0];             // chunk0, v=0..3
        const v8h q1 = ldsB8[g5 + 1];             // chunk0, v=4..7
        const v8h q2 = ldsB8[g5 + 2];             // chunk1, v=0..3
        const v8h q3 = ldsB8[g5 + 3];             // chunk1, v=4..7
        const v16h b0 = __builtin_shufflevector(q0, q1, 0, 1, 2, 3, 4, 5, 6, 7,
                                                8, 9, 10, 11, 12, 13, 14, 15);
        const v16h b1 = __builtin_shufflevector(q2, q3, 0, 1, 2, 3, 4, 5, 6, 7,
                                                8, 9, 10, 11, 12, 13, 14, 15);

        v8f c = {};
        c = __builtin_amdgcn_wmma_f32_16x16x32_f16(false, a0, false, b0, (short)0, c, false, false);
        c = __builtin_amdgcn_wmma_f32_16x16x32_f16(false, a1, false, b1, (short)0, c, false, false);

        const float bias = bx[n];
        const float wrn  = Wr[n];
#pragma unroll
        for (int v = 0; v < 8; ++v)
            acc[v] += tanh_fast(c[v] + bias) * wrn;
    }

    // ---------- reduce the N dimension across the 16 lanes of each half ----------
#pragma unroll
    for (int v = 0; v < 8; ++v) {
        acc[v] += __shfl_xor(acc[v], 1, 32);
        acc[v] += __shfl_xor(acc[v], 2, 32);
        acc[v] += __shfl_xor(acc[v], 4, 32);
        acc[v] += __shfl_xor(acc[v], 8, 32);
    }

    // ---------- sigmoid + softmax over the 4 rows of each window ----------
    const float brv = br[0];
    float wt[8];
#pragma unroll
    for (int v = 0; v < 8; ++v)
        wt[v] = sigmoid_fast(acc[v] + brv);       // sigmoid(logit)
#pragma unroll
    for (int g = 0; g < 2; ++g) {
        const float mx = fmaxf(fmaxf(wt[4 * g], wt[4 * g + 1]),
                               fmaxf(wt[4 * g + 2], wt[4 * g + 3]));
        float s = 0.0f;
#pragma unroll
        for (int v = 0; v < 4; ++v) {
            wt[4 * g + v] = __expf(wt[4 * g + v] - mx);
            s += wt[4 * g + v];
        }
        const float inv = __builtin_amdgcn_rcpf(s);
#pragma unroll
        for (int v = 0; v < 4; ++v) wt[4 * g + v] *= inv;
    }
    // lane group g = lane>>4 now holds weights for windows (2g, 2g+1) in wt[0..3], wt[4..7]

    // ---------- fp32 weighted sum: each lane does 2 channels of all 4 windows ----------
    const int d = lane * 2;
#pragma unroll
    for (int j = 0; j < 4; ++j) {
        const long wj = winBase + j;
        const int  pj = (int)(wj & 63);
        const int  qj = (int)((wj >> 6) & 63);
        const int  bj = (int)(wj >> 12);
        const float* base = x + (((long)bj * RR + 2 * qj) * CC + 2 * pj) * DIMD + d;
        const int srcLane = (j >> 1) * 16;
        float2 o; o.x = 0.0f; o.y = 0.0f;
#pragma unroll
        for (int e = 0; e < 4; ++e) {
            const float wgt = __shfl(wt[(j & 1) * 4 + e], srcLane, 32);
            const float2 xi = *(const float2*)(base + (((e >> 1) * CC) + (e & 1)) * DIMD);
            o.x += xi.x * wgt;
            o.y += xi.y * wgt;
        }
        float* op = out + (((long)bj * SRR + qj) * SCC + pj) * DIMD + d;
        *(float2*)op = o;
    }
}

extern "C" void kernel_launch(void* const* d_in, const int* in_sizes, int n_in,
                              void* d_out, int out_size, void* d_ws, size_t ws_size,
                              hipStream_t stream) {
    const float* x  = (const float*)d_in[0];
    const float* Wx = (const float*)d_in[1];
    const float* bx = (const float*)d_in[2];
    const float* Wr = (const float*)d_in[3];
    const float* br = (const float*)d_in[4];
    // d_in[5], d_in[6] are kernel_size=2 / stride=2 scalars (hardcoded in the kernel)
    float* out = (float*)d_out;

    // 131072 windows / 4 per wave / 8 waves per block = 4096 blocks, exact fit
    softpool2d_wmma_kernel<<<4096, 256, 0, stream>>>(x, Wx, bx, Wr, br, out);
}
